// AdaptiveMSSampler_35407710388293
// MI455X (gfx1250) — matrix operand order repaired and checked by reading, not compiled, not run on hardware
//
#include <hip/hip_runtime.h>
#include <hip/hip_bf16.h>
#include <math.h>

#define HH 512
#define WWID 512
#define HWPIX (HH*WWID)
#define BB 2
#define NPTS 131072
#define CST 48
#define KC 160   // K-chunk staged in LDS (multiple of 32)

typedef __attribute__((ext_vector_type(16))) __bf16 v16bf;
typedef __attribute__((ext_vector_type(8)))  float  v8f;
typedef unsigned int v4u __attribute__((ext_vector_type(4)));
typedef int          v4i __attribute__((ext_vector_type(4)));
typedef int          v8i __attribute__((ext_vector_type(8)));

#define HAS_TDM (__has_builtin(__builtin_amdgcn_tensor_load_to_lds) && \
                 __has_builtin(__builtin_amdgcn_s_wait_tensorcnt))

__device__ __forceinline__ unsigned short f2bf(float f) {
    unsigned int u = __float_as_uint(f);
    u += 0x7fffu + ((u >> 16) & 1u);           // round-to-nearest-even
    return (unsigned short)(u >> 16);
}
__device__ __forceinline__ float bf2f(unsigned short h) {
    return __uint_as_float(((unsigned int)h) << 16);
}

// --------------------------- weight repack: OIHW f32 -> [Cout][KPAD] bf16
// k = tap*Cin + cin, zero-padded to KPAD. Produces a contiguous 2D tile the
// TDM can DMA straight into LDS.
__global__ void repack_weights(const float* __restrict__ w,
                               unsigned short* __restrict__ dst,
                               int Cin, int Cout, int Kpad) {
    int idx = blockIdx.x * 256 + threadIdx.x;
    int total = Cout * Kpad;
    if (idx >= total) return;
    int o = idx / Kpad;
    int k = idx - o * Kpad;
    int Ktot = Cin * 9;
    unsigned short v = 0;
    if (k < Ktot) {
        int cin = k % Cin;
        int tap = k / Cin;
        v = f2bf(w[((size_t)o * Cin + cin) * 9 + tap]);
    }
    dst[idx] = v;
}

// ------------------------------------------------- layer0: conv3x3 3->16 f32
__global__ void conv0_kernel(const float* __restrict__ x0,
                             const float* __restrict__ x1,
                             const float* __restrict__ x2,
                             const float* __restrict__ w,
                             const float* __restrict__ bias,
                             const float* __restrict__ mask,
                             unsigned short* __restrict__ dst) {
    __shared__ float sw[16 * 27];
    __shared__ float sb[16];
    int tid = threadIdx.x;
    for (int i = tid; i < 432; i += blockDim.x) sw[i] = w[i];
    if (tid < 16) sb[tid] = bias[tid];
    __syncthreads();

    int z = blockIdx.z;
    int b = z / 3, img = z % 3;
    const float* x = (img == 0) ? x0 : ((img == 1) ? x1 : x2);
    int p = blockIdx.x * blockDim.x + tid;
    int y = p >> 9, xx = p & (WWID - 1);

    float acc[16];
#pragma unroll
    for (int o = 0; o < 16; ++o) acc[o] = sb[o];

    for (int c = 0; c < 3; ++c) {
        const float* xb = x + ((size_t)(b * 3 + c)) * HWPIX;
#pragma unroll
        for (int ky = 0; ky < 3; ++ky) {
            int gy = y + ky - 1;
#pragma unroll
            for (int kx = 0; kx < 3; ++kx) {
                int gx = xx + kx - 1;
                float v = 0.f;
                if (gy >= 0 && gy < HH && gx >= 0 && gx < WWID)
                    v = xb[gy * WWID + gx];
#pragma unroll
                for (int o = 0; o < 16; ++o)
                    acc[o] = fmaf(v, sw[((o * 3 + c) * 3 + ky) * 3 + kx], acc[o]);
            }
        }
    }
    float mk = mask[p];
    unsigned short* dp = dst + ((size_t)b * HWPIX + p) * CST + img * 16;
#pragma unroll
    for (int o = 0; o < 16; ++o) {
        float v = acc[o];
        v = v > 0.f ? v : 0.f;
        v *= mk;
        dp[o] = f2bf(v);
    }
}

// -------------------------------------- generic WMMA implicit-GEMM conv 3x3
// M tile = 64 pixels (4 waves x 16), N = NOB*16 output channels per wave
// (NOB accumulators reuse each A fragment), K = 9*CIN padded to KPAD.
// Weights arrive in LDS via Tensor Data Mover (fallback: cooperative copy).
template <int CIN, int NOB>
__global__ void __launch_bounds__(128)
wmma_conv3x3(const unsigned short* __restrict__ src,
             unsigned short* __restrict__ dst,
             const unsigned short* __restrict__ wpad,  // [NOB*16][KPAD] bf16
             const float* __restrict__ bias,
             const float* __restrict__ mask,
             int imgCount, int doRelu) {
    constexpr int KTOT = CIN * 9;
    constexpr int KPAD = (KTOT + 31) & ~31;
    constexpr int NR   = NOB * 16;

    __shared__ unsigned short Alds[64][KC];
    __shared__ unsigned short Blds[NR][KPAD];

    int tid  = threadIdx.x;
    int lane = tid & 31;
    int wv   = tid >> 5;
    int z    = blockIdx.z;
    int b    = z / imgCount;
    int img  = z - b * imgCount;
    int chOff = (imgCount > 1) ? img * 16 : 0;
    int p0    = blockIdx.x * 64;

    __builtin_prefetch(src + ((size_t)b * HWPIX + p0) * CST, 0, 1);

    // ---- stage B (weights) into LDS via TDM -------------------------------
#if HAS_TDM
    if (wv == 0) {   // one wave issues the DMA; EXEC is all-ones in wave 0
        unsigned lds_off = (unsigned)(size_t)(void*)&Blds[0][0]; // LDS aperture: low 32 bits = LDS offset
        unsigned long long ga = (unsigned long long)(size_t)wpad;
        v4u g0;
        g0.x = 1u;                                        // count=1, user D#
        g0.y = lds_off;                                   // lds_addr
        g0.z = (unsigned)(ga & 0xFFFFFFFFu);              // global_addr[31:0]
        g0.w = (unsigned)((ga >> 32) & 0x01FFFFFFu) | 0x80000000u; // addr[56:32] | type=2
        v8i g1;
        g1[0] = 0x00010000;            // workgroup_mask=0, data_size=1 (2B)
        g1[1] = (int)(KPAD << 16);     // tensor_dim0[15:0]
        g1[2] = (int)(NR << 16);       // tensor_dim0[31:16]=0 | tensor_dim1[15:0]
        g1[3] = (int)(KPAD << 16);     // tensor_dim1[31:16]=0 | tile_dim0
        g1[4] = (int)NR;               // tile_dim1 | tile_dim2=0
        g1[5] = (int)KPAD;             // tensor_dim0_stride[31:0]
        g1[6] = 0;                     // stride0[47:32] | stride1[15:0]
        g1[7] = 0;
        v4i gz = {0, 0, 0, 0};
#if __clang_major__ >= 23
        v8i gz8 = {0, 0, 0, 0, 0, 0, 0, 0};
        __builtin_amdgcn_tensor_load_to_lds(g0, g1, gz, gz, gz8, 0);
#else
        __builtin_amdgcn_tensor_load_to_lds(g0, g1, gz, gz, 0);
#endif
        __builtin_amdgcn_s_wait_tensorcnt(0);
    }
#else
    {
        unsigned short* bflat = &Blds[0][0];
        for (int idx = tid; idx < NR * KPAD; idx += 128) bflat[idx] = wpad[idx];
    }
#endif
    __syncthreads();

    v8f acc[NOB];
#pragma unroll
    for (int ob = 0; ob < NOB; ++ob)
        acc[ob] = (v8f){0.f, 0.f, 0.f, 0.f, 0.f, 0.f, 0.f, 0.f};

    int mloc = lane & 15;
    int half = lane >> 4;

#pragma unroll
    for (int kc = 0; kc < KPAD; kc += KC) {
        // stage A chunk: [pixel 0..63][k in chunk], k = tap*CIN + cin
        for (int idx = tid; idx < 64 * KC; idx += 128) {
            int pl = idx / KC;
            int kk = idx - pl * KC;
            int kg = kc + kk;
            unsigned short v = 0;
            if (kg < KTOT) {
                int cin = kg % CIN;
                int tap = kg / CIN;
                int ky = tap / 3, kx = tap - ky * 3;
                int p = p0 + pl;
                int y = p >> 9, x = p & (WWID - 1);
                int gy = y + ky - 1, gx = x + kx - 1;
                if (gy >= 0 && gy < HH && gx >= 0 && gx < WWID)
                    v = src[((size_t)b * HWPIX + gy * WWID + gx) * CST + chOff + cin];
            }
            Alds[pl][kk] = v;
        }
        __syncthreads();

        constexpr int FULL = KC;
        int rem = (KPAD - kc) < FULL ? (KPAD - kc) : FULL;
#pragma unroll
        for (int kb = 0; kb < FULL; kb += 32) {
            if (kb >= rem) break;
            union { v16bf v; unsigned int u[8]; } af;
            // A frag (16x32): lanes 0-15 K{0-15}, lanes 16-31 K{16-31}
            const unsigned short* arow = &Alds[wv * 16 + mloc][kb + half * 16];
#pragma unroll
            for (int j = 0; j < 4; ++j) {
                af.u[j]     = *(const unsigned int*)(arow + 2 * j);
                af.u[4 + j] = *(const unsigned int*)(arow + 8 + 2 * j);
            }
#pragma unroll
            for (int ob = 0; ob < NOB; ++ob) {
                union { v16bf v; unsigned int u[8]; } bw;
                // B frag (32x16): lanes 0-15 pairs kb+2v, lanes 16-31 kb+16+2v
                const unsigned short* brow = &Blds[ob * 16 + mloc][kc + kb + half * 16];
#pragma unroll
                for (int j = 0; j < 8; ++j)
                    bw.u[j] = *(const unsigned int*)(brow + 2 * j);
                acc[ob] = __builtin_amdgcn_wmma_f32_16x16x32_bf16(
                    false, af.v, false, bw.v, (short)0, acc[ob], false, false);
            }
        }
        __syncthreads();
    }

    // epilogue: C/D layout VGPR r -> M=r (lanes 0-15) / M=r+8 (lanes 16-31)
    int mbase = wv * 16 + half * 8;
#pragma unroll
    for (int ob = 0; ob < NOB; ++ob) {
        int o_local = ob * 16 + mloc;
        float bv = bias[o_local];
#pragma unroll
        for (int r = 0; r < 8; ++r) {
            int p = p0 + mbase + r;
            float v = acc[ob][r] + bv;
            if (doRelu) v = v > 0.f ? v : 0.f;
            v *= mask[p];
            dst[((size_t)b * HWPIX + p) * CST + chOff + o_local] = f2bf(v);
        }
    }
}

// ------------------- gather + log-Jacobian + pointwise MLP + softmax scatter
// 64 threads/block, activations ping-pong through LDS (no register arrays).
__global__ void __launch_bounds__(64)
sampler_kernel(const unsigned short* __restrict__ feat,
               const int* __restrict__ yi,
               const float* __restrict__ mmat,
               const float* __restrict__ w1, const float* __restrict__ w2,
               const float* __restrict__ w3, const float* __restrict__ w4,
               const float* __restrict__ b1, const float* __restrict__ b2,
               const float* __restrict__ b3, const float* __restrict__ b4,
               float* __restrict__ wgrid) {
    __shared__ float sW1[48 * 49], sW2[48 * 48], sW3[48 * 48], sW4[3 * 48];
    __shared__ float sB1[48], sB2[48], sB3[48], sB4[3];
    __shared__ float fbuf[64][51];   // stride 51: gcd(51,64)=1, conflict-free
    __shared__ float hbuf[64][51];
    int tid = threadIdx.x;
    for (int i = tid; i < 48 * 49; i += 64) sW1[i] = w1[i];
    for (int i = tid; i < 48 * 48; i += 64) sW2[i] = w2[i];
    for (int i = tid; i < 48 * 48; i += 64) sW3[i] = w3[i];
    for (int i = tid; i < 3 * 48;  i += 64) sW4[i] = w4[i];
    if (tid < 48) { sB1[tid] = b1[tid]; sB2[tid] = b2[tid]; sB3[tid] = b3[tid]; }
    if (tid < 3) sB4[tid] = b4[tid];

    int gi = blockIdx.x * 64 + tid;      // 0 .. B*N-1
    int b = gi / NPTS;
    int n = gi - b * NPTS;
    int pix = yi[n];

    // log |det(m)| / |den|^3 density term
    float m00 = mmat[0], m01 = mmat[1], m02 = mmat[2];
    float m10 = mmat[3], m11 = mmat[4], m12 = mmat[5];
    float m20 = mmat[6], m21 = mmat[7], m22 = mmat[8];
    float det = m00 * (m11 * m22 - m12 * m21)
              - m01 * (m10 * m22 - m12 * m20)
              + m02 * (m10 * m21 - m11 * m20);
    float fy = (float)(pix >> 9);
    float fx = (float)(pix & (WWID - 1));
    float den = m20 * fx + m21 * fy + m22;
    float ad = fabsf(den);
    float dsda = fabsf(det) / (ad * ad * ad + 1e-12f);

    const unsigned short* fp = feat + ((size_t)b * HWPIX + pix) * CST;
#pragma unroll
    for (int c = 0; c < 48; ++c) fbuf[tid][c] = bf2f(fp[c]);
    fbuf[tid][48] = logf(dsda + 1e-8f);
    __syncthreads();

    for (int o = 0; o < 48; ++o) {           // layer 1: 49 -> 48, ReLU
        float s = sB1[o];
        for (int c = 0; c < 49; ++c) s = fmaf(sW1[o * 49 + c], fbuf[tid][c], s);
        hbuf[tid][o] = s > 0.f ? s : 0.f;
    }
    for (int o = 0; o < 48; ++o) {           // layer 2: 48 -> 48, ReLU
        float s = sB2[o];
        for (int c = 0; c < 48; ++c) s = fmaf(sW2[o * 48 + c], hbuf[tid][c], s);
        fbuf[tid][o] = s > 0.f ? s : 0.f;
    }
    for (int o = 0; o < 48; ++o) {           // layer 3: 48 -> 48, ReLU
        float s = sB3[o];
        for (int c = 0; c < 48; ++c) s = fmaf(sW3[o * 48 + c], fbuf[tid][c], s);
        hbuf[tid][o] = s > 0.f ? s : 0.f;
    }
    float o3[3];
    float mx = -3.4e38f;
#pragma unroll
    for (int k = 0; k < 3; ++k) {            // layer 4: 48 -> 3
        float s = sB4[k];
        for (int c = 0; c < 48; ++c) s = fmaf(sW4[k * 48 + c], hbuf[tid][c], s);
        o3[k] = s;
        mx = s > mx ? s : mx;
    }
    float e0 = expf(o3[0] - mx), e1 = expf(o3[1] - mx), e2 = expf(o3[2] - mx);
    float inv = 1.f / (e0 + e1 + e2);
    // duplicates in yi produce identical values -> benign racing scatter
    wgrid[((size_t)b * 3 + 0) * HWPIX + pix] = e0 * inv;
    wgrid[((size_t)b * 3 + 1) * HWPIX + pix] = e1 * inv;
    wgrid[((size_t)b * 3 + 2) * HWPIX + pix] = e2 * inv;
}

// ------------------------------------------------- out = sum_k w_k * x_k
__global__ void combine_kernel(const float* __restrict__ wgrid,
                               const float* __restrict__ x0,
                               const float* __restrict__ x1,
                               const float* __restrict__ x2,
                               float* __restrict__ out) {
    int idx = blockIdx.x * 256 + threadIdx.x;   // B*3*HW
    int b = idx / (3 * HWPIX);
    int r = idx - b * 3 * HWPIX;
    int c = r / HWPIX;
    int p = r - c * HWPIX;
    size_t xo = ((size_t)b * 3 + c) * HWPIX + p;
    float w0 = wgrid[((size_t)b * 3 + 0) * HWPIX + p];
    float w1 = wgrid[((size_t)b * 3 + 1) * HWPIX + p];
    float w2 = wgrid[((size_t)b * 3 + 2) * HWPIX + p];
    out[idx] = w0 * x0[xo] + w1 * x1[xo] + w2 * x2[xo];
}

extern "C" void kernel_launch(void* const* d_in, const int* in_sizes, int n_in,
                              void* d_out, int out_size, void* d_ws, size_t ws_size,
                              hipStream_t stream) {
    const float* x0   = (const float*)d_in[0];
    const float* x1   = (const float*)d_in[1];
    const float* x2   = (const float*)d_in[2];
    const float* mask = (const float*)d_in[3];
    const int*   yi   = (const int*)d_in[4];
    const float* mmat = (const float*)d_in[5];
    const float* sw0  = (const float*)d_in[6];
    const float* sw1  = (const float*)d_in[7];
    const float* sw2  = (const float*)d_in[8];
    const float* sb0  = (const float*)d_in[9];
    const float* sb1  = (const float*)d_in[10];
    const float* sb2  = (const float*)d_in[11];
    const float* fw0  = (const float*)d_in[12];
    const float* fw1  = (const float*)d_in[13];
    const float* fw2  = (const float*)d_in[14];
    const float* fb0  = (const float*)d_in[15];
    const float* fb1  = (const float*)d_in[16];
    const float* fb2  = (const float*)d_in[17];
    const float* pw0  = (const float*)d_in[18];
    const float* pw1  = (const float*)d_in[19];
    const float* pw2  = (const float*)d_in[20];
    const float* pw3  = (const float*)d_in[21];
    const float* pb0  = (const float*)d_in[22];
    const float* pb1  = (const float*)d_in[23];
    const float* pb2  = (const float*)d_in[24];
    const float* pb3  = (const float*)d_in[25];

    char* ws = (char*)d_ws;
    unsigned short* actA = (unsigned short*)ws; ws += (size_t)BB * HWPIX * CST * 2;
    unsigned short* actB = (unsigned short*)ws; ws += (size_t)BB * HWPIX * CST * 2;
    float* wgrid         = (float*)ws;          ws += (size_t)BB * 3 * HWPIX * 4;
    unsigned short* wp_s1 = (unsigned short*)ws; ws += 16 * 160 * 2;   // [16][160]
    unsigned short* wp_s2 = (unsigned short*)ws; ws += 16 * 160 * 2;
    unsigned short* wp_f0 = (unsigned short*)ws; ws += 48 * 448 * 2;   // [48][448]
    unsigned short* wp_f1 = (unsigned short*)ws; ws += 48 * 448 * 2;
    unsigned short* wp_f2 = (unsigned short*)ws; ws += 48 * 448 * 2;

    // 1) weights -> padded K-major bf16 tiles (TDM-ready)
    repack_weights<<<10, 256, 0, stream>>>(sw1, wp_s1, 16, 16, 160);
    repack_weights<<<10, 256, 0, stream>>>(sw2, wp_s2, 16, 16, 160);
    repack_weights<<<84, 256, 0, stream>>>(fw0, wp_f0, 48, 48, 448);
    repack_weights<<<84, 256, 0, stream>>>(fw1, wp_f1, 48, 48, 448);
    repack_weights<<<84, 256, 0, stream>>>(fw2, wp_f2, 48, 48, 448);

    // 2) shared layer 0 (3->16, fp32 direct) for 3 images x B
    conv0_kernel<<<dim3(HWPIX / 256, 1, BB * 3), 256, 0, stream>>>(
        x0, x1, x2, sw0, sb0, mask, actA);

    // 3) WMMA conv stack
    dim3 gS(HWPIX / 64, 1, BB * 3);   // per-image 16->16
    wmma_conv3x3<16, 1><<<gS, 128, 0, stream>>>(actA, actB, wp_s1, sb1, mask, 3, 1);
    wmma_conv3x3<16, 1><<<gS, 128, 0, stream>>>(actB, actA, wp_s2, sb2, mask, 3, 0);
    dim3 gF(HWPIX / 64, 1, BB);       // 48->48, 3 accumulators/wave
    wmma_conv3x3<48, 3><<<gF, 128, 0, stream>>>(actA, actB, wp_f0, fb0, mask, 1, 1);
    wmma_conv3x3<48, 3><<<gF, 128, 0, stream>>>(actB, actA, wp_f1, fb1, mask, 1, 1);
    wmma_conv3x3<48, 3><<<gF, 128, 0, stream>>>(actA, actB, wp_f2, fb2, mask, 1, 0);

    // 4) sampler MLP + softmax scatter (wgrid zeroed first, every call)
    hipMemsetAsync(wgrid, 0, (size_t)BB * 3 * HWPIX * sizeof(float), stream);
    sampler_kernel<<<(BB * NPTS) / 64, 64, 0, stream>>>(
        actB, yi, mmat, pw0, pw1, pw2, pw3, pb0, pb1, pb2, pb3, wgrid);

    // 5) final combine
    combine_kernel<<<(BB * 3 * HWPIX) / 256, 256, 0, stream>>>(
        wgrid, x0, x1, x2, (float*)d_out);

    (void)in_sizes; (void)n_in; (void)out_size; (void)ws_size;
}
